// GAU_22660247454339
// MI455X (gfx1250) — compile-verified, matrix operand order checked by reading, not run on hardware
//
#include <hip/hip_runtime.h>
#include <hip/hip_bf16.h>

typedef __bf16 bf16;
typedef __attribute__((ext_vector_type(16))) __bf16 bf16x16;
typedef __attribute__((ext_vector_type(8)))  float  floatx8;

#define DIM  1024
#define QKD  128
#define HID  2048
#define SEQ  4096
#define NTOK 8192   // B * N
#define BM   128
#define BN   128
#define BK   64     // two WMMA K-steps per staged tile

union FragU { uint4 u[2]; bf16x16 v; };

// ---------------------------------------------------------------------------
// Async-pipelined WMMA main loop: C(128x128) += A(128xK) * BT(128xK)^T
// A row-major [M][K], BT row-major [N][K].
// 128 threads = 4 waves in a 2(M) x 2(N) grid; each wave owns a 64x64
// sub-tile = 4x4 fragments of 16x16 (fp32 accum).
// Global->LDS staging uses gfx1250 global_load_async_to_lds_b128 with
// double-buffered LDS and ASYNCcnt-based software pipelining; the last
// iteration is peeled so the steady-state loop is branch-free.
// ---------------------------------------------------------------------------
__device__ __forceinline__ void gemm_mainloop(
    const bf16* __restrict__ A, int lda,
    const bf16* __restrict__ BT, int ldb,
    int K, bf16 (*As)[BM * BK], bf16 (*Bs)[BN * BK], floatx8 acc[4][4])
{
  const int tid   = threadIdx.x;     // 0..127
  const int lane  = tid & 31;
  const int wave  = tid >> 5;        // 0..3
  const int waveM = wave >> 1;       // 0..1
  const int waveN = wave & 1;        // 0..1
  const int row16 = lane & 15;
  const int khalf = lane >> 4;       // K-half select per WMMA 16-bit layout

  // LDS byte addresses (low 32 bits of the generic pointer = LDS offset)
  const unsigned aLds0  = (unsigned)(size_t)(&As[0][0]);
  const unsigned bLds0  = (unsigned)(size_t)(&Bs[0][0]);
  const unsigned ldsRow = (unsigned)tid * (BK * 2);          // 128 B per tile row
  const unsigned gRowA  = (unsigned)tid * (unsigned)lda * 2; // global byte offsets
  const unsigned gRowB  = (unsigned)tid * (unsigned)ldb * 2;

  // Each thread streams one full 64-element row of each tile: 8 x b128 each.
  auto fill = [&](int buf, int k0) {
    const unsigned aL = aLds0 + (unsigned)buf * (BM * BK * 2) + ldsRow;
    const unsigned bL = bLds0 + (unsigned)buf * (BN * BK * 2) + ldsRow;
    const unsigned ga = gRowA + (unsigned)k0 * 2;
    const unsigned gb = gRowB + (unsigned)k0 * 2;
#pragma unroll
    for (int s = 0; s < (BK * 2) / 16; ++s) {          // 8 segments of 16 B
      asm volatile("global_load_async_to_lds_b128 %0, %1, %2"
                   :: "v"(aL + (unsigned)s * 16u), "v"(ga + (unsigned)s * 16u),
                      "s"(A) : "memory");
      asm volatile("global_load_async_to_lds_b128 %0, %1, %2"
                   :: "v"(bL + (unsigned)s * 16u), "v"(gb + (unsigned)s * 16u),
                      "s"(BT) : "memory");
    }
  };

  auto compute = [&](int buf) {
    const bf16* as = As[buf];
    const bf16* bs = Bs[buf];
#pragma unroll
    for (int ks = 0; ks < BK / 32; ++ks) {             // 2 WMMA K-steps
      bf16x16 bfrag[4];
#pragma unroll
      for (int nf = 0; nf < 4; ++nf) {
        const int col = waveN * 64 + nf * 16 + row16;
        FragU f;
        f.u[0] = *(const uint4*)(bs + col * BK + ks * 32 + 8 * khalf);
        f.u[1] = *(const uint4*)(bs + col * BK + ks * 32 + 16 + 8 * khalf);
        bfrag[nf] = f.v;
      }
#pragma unroll
      for (int mf = 0; mf < 4; ++mf) {
        const int row = waveM * 64 + mf * 16 + row16;
        FragU f;
        f.u[0] = *(const uint4*)(as + row * BK + ks * 32 + 8 * khalf);
        f.u[1] = *(const uint4*)(as + row * BK + ks * 32 + 16 + 8 * khalf);
#pragma unroll
        for (int nf = 0; nf < 4; ++nf)
          acc[mf][nf] = __builtin_amdgcn_wmma_f32_16x16x32_bf16(
              false, f.v, false, bfrag[nf], (short)0, acc[mf][nf], false, false);
      }
    }
  };

  fill(0, 0);                                   // prologue: prefetch tile 0
  const int iters = K / BK;
  for (int it = 0; it < iters - 1; ++it) {      // branch-free steady state
    fill((it + 1) & 1, (it + 1) * BK);          // prefetch next tile
    asm volatile("s_wait_asynccnt 16" ::: "memory");  // current tile landed
    __syncthreads();
    compute(it & 1);
    __syncthreads();                            // protect buf from next prefetch
  }
  asm volatile("s_wait_asynccnt 0" ::: "memory");     // peeled tail
  __syncthreads();
  compute((iters - 1) & 1);
}

#define GEMM_SETUP                                                     \
  __shared__ __align__(16) bf16 As[2][BM * BK];                        \
  __shared__ __align__(16) bf16 Bs[2][BN * BK];                        \
  floatx8 acc[4][4];                                                   \
  for (int i = 0; i < 4; ++i)                                          \
    for (int j = 0; j < 4; ++j)                                        \
      for (int e = 0; e < 8; ++e) acc[i][j][e] = 0.0f;

#define EPILOGUE_IDX                                                   \
  const int lane  = threadIdx.x & 31;                                  \
  const int wave  = threadIdx.x >> 5;                                  \
  const int mBase = blockIdx.y * BM + (wave >> 1) * 64;                \
  const int nBase = blockIdx.x * BN + (wave & 1) * 64;

__device__ __forceinline__ float silu(float t) {
  // fast SiLU: native exp2-based exp + hardware reciprocal (bf16-rounded out)
  const float e = __expf(-t);
  return t * __builtin_amdgcn_rcpf(1.0f + e);
}

// ---------------------------------------------------------------------------
// GEMM 1: normed @ Wh  (8192 x 4096 x 1024), SiLU, split into vT and gate
// ---------------------------------------------------------------------------
__global__ void __launch_bounds__(128) k_gemm_hv(
    const bf16* __restrict__ nrm, const bf16* __restrict__ WhT,
    const float* __restrict__ bh, bf16* __restrict__ vT, bf16* __restrict__ gate)
{
  GEMM_SETUP
  gemm_mainloop(nrm + (size_t)blockIdx.y * BM * DIM, DIM,
                WhT + (size_t)blockIdx.x * BN * DIM, DIM, DIM, As, Bs, acc);
  EPILOGUE_IDX
#pragma unroll
  for (int mf = 0; mf < 4; ++mf)
#pragma unroll
    for (int nf = 0; nf < 4; ++nf)
#pragma unroll
      for (int r = 0; r < 8; ++r) {
        const int m = mBase + mf * 16 + r + 8 * (lane >> 4);
        const int n = nBase + nf * 16 + (lane & 15);
        float t = silu(acc[mf][nf][r] + bh[n]);
        if (n < HID) vT[(size_t)n * NTOK + m] = (bf16)t;           // transposed v
        else         gate[(size_t)m * HID + (n - HID)] = (bf16)t;  // gate
      }
}

// ---------------------------------------------------------------------------
// GEMM 2: normed @ Wqk (8192 x 128 x 1024), SiLU, q/k with gamma/beta
// ---------------------------------------------------------------------------
__global__ void __launch_bounds__(128) k_gemm_qk(
    const bf16* __restrict__ nrm, const bf16* __restrict__ WqkT,
    const float* __restrict__ bqk, const float* __restrict__ gamma,
    const float* __restrict__ beta, bf16* __restrict__ q, bf16* __restrict__ k)
{
  GEMM_SETUP
  gemm_mainloop(nrm + (size_t)blockIdx.y * BM * DIM, DIM,
                WqkT + (size_t)blockIdx.x * BN * DIM, DIM, DIM, As, Bs, acc);
  EPILOGUE_IDX
#pragma unroll
  for (int mf = 0; mf < 4; ++mf)
#pragma unroll
    for (int nf = 0; nf < 4; ++nf)
#pragma unroll
      for (int r = 0; r < 8; ++r) {
        const int m = mBase + mf * 16 + r + 8 * (lane >> 4);
        const int n = nBase + nf * 16 + (lane & 15);
        float t = silu(acc[mf][nf][r] + bqk[n]);
        q[(size_t)m * QKD + n] = (bf16)(t * gamma[n]       + beta[n]);
        k[(size_t)m * QKD + n] = (bf16)(t * gamma[QKD + n] + beta[QKD + n]);
      }
}

// ---------------------------------------------------------------------------
// GEMM 3: sim = q @ k^T / n, relu^2, causal mask  (per batch, 4096x4096x128)
// Tiles strictly above the causal diagonal skip the GEMM and store zeros.
// ---------------------------------------------------------------------------
__global__ void __launch_bounds__(128) k_gemm_sim(
    const bf16* __restrict__ qb, const bf16* __restrict__ kb,
    bf16* __restrict__ attn)
{
  if (blockIdx.x * BN > blockIdx.y * BM + (BM - 1)) {   // fully masked tile
    bf16* dst = attn + ((size_t)(blockIdx.y * BM) + threadIdx.x) * SEQ
                     + (size_t)blockIdx.x * BN;
    const uint4 z = make_uint4(0u, 0u, 0u, 0u);
#pragma unroll
    for (int s = 0; s < 16; ++s) ((uint4*)dst)[s] = z;  // 128 bf16 per row
    return;
  }
  GEMM_SETUP
  gemm_mainloop(qb + (size_t)blockIdx.y * BM * QKD, QKD,
                kb + (size_t)blockIdx.x * BN * QKD, QKD, QKD, As, Bs, acc);
  EPILOGUE_IDX
#pragma unroll
  for (int mf = 0; mf < 4; ++mf)
#pragma unroll
    for (int nf = 0; nf < 4; ++nf)
#pragma unroll
      for (int r = 0; r < 8; ++r) {
        const int i = mBase + mf * 16 + r + 8 * (lane >> 4);
        const int j = nBase + nf * 16 + (lane & 15);
        float s = acc[mf][nf][r] * (1.0f / (float)SEQ);
        float a = (j <= i && s > 0.0f) ? s * s : 0.0f;
        attn[(size_t)i * SEQ + j] = (bf16)a;
      }
}

// ---------------------------------------------------------------------------
// GEMM 4: out = (attn @ v) * gate  (per batch, 4096 x 2048 x 4096)
// ---------------------------------------------------------------------------
__global__ void __launch_bounds__(128) k_gemm_av(
    const bf16* __restrict__ attn, const bf16* __restrict__ vT,
    const bf16* __restrict__ gate, bf16* __restrict__ u)
{
  GEMM_SETUP
  gemm_mainloop(attn + (size_t)blockIdx.y * BM * SEQ, SEQ,
                vT + (size_t)blockIdx.x * BN * NTOK, NTOK, SEQ, As, Bs, acc);
  EPILOGUE_IDX
#pragma unroll
  for (int mf = 0; mf < 4; ++mf)
#pragma unroll
    for (int nf = 0; nf < 4; ++nf)
#pragma unroll
      for (int r = 0; r < 8; ++r) {
        const int m = mBase + mf * 16 + r + 8 * (lane >> 4);
        const int n = nBase + nf * 16 + (lane & 15);
        float g = (float)gate[(size_t)m * HID + n];
        u[(size_t)m * HID + n] = (bf16)(acc[mf][nf][r] * g);
      }
}

// ---------------------------------------------------------------------------
// GEMM 5: out = u @ Wo + bo + x  (8192 x 1024 x 2048), fp32 output
// ---------------------------------------------------------------------------
__global__ void __launch_bounds__(128) k_gemm_out(
    const bf16* __restrict__ u, const bf16* __restrict__ WoT,
    const float* __restrict__ bo, const float* __restrict__ xres,
    float* __restrict__ out)
{
  GEMM_SETUP
  gemm_mainloop(u + (size_t)blockIdx.y * BM * HID, HID,
                WoT + (size_t)blockIdx.x * BN * HID, HID, HID, As, Bs, acc);
  EPILOGUE_IDX
#pragma unroll
  for (int mf = 0; mf < 4; ++mf)
#pragma unroll
    for (int nf = 0; nf < 4; ++nf)
#pragma unroll
      for (int r = 0; r < 8; ++r) {
        const int m = mBase + mf * 16 + r + 8 * (lane >> 4);
        const int n = nBase + nf * 16 + (lane & 15);
        out[(size_t)m * DIM + n] =
            acc[mf][nf][r] + bo[n] + xres[(size_t)m * DIM + n];
      }
}

// ---------------------------------------------------------------------------
// LayerNorm: one 256-thread block per row (DIM = 1024 = 4 floats/thread)
// ---------------------------------------------------------------------------
__global__ void __launch_bounds__(256) k_layernorm(
    const float* __restrict__ x, const float* __restrict__ g,
    const float* __restrict__ b, bf16* __restrict__ nrm)
{
  const int row = blockIdx.x;
  const int tid = threadIdx.x;
  const float4 v = ((const float4*)(x + (size_t)row * DIM))[tid];
  __shared__ float r1[256], r2[256];
  r1[tid] = v.x + v.y + v.z + v.w;
  r2[tid] = v.x * v.x + v.y * v.y + v.z * v.z + v.w * v.w;
  __syncthreads();
  for (int o = 128; o > 0; o >>= 1) {
    if (tid < o) { r1[tid] += r1[tid + o]; r2[tid] += r2[tid + o]; }
    __syncthreads();
  }
  const float mu   = r1[0] * (1.0f / DIM);
  const float var  = r2[0] * (1.0f / DIM) - mu * mu;
  const float rstd = rsqrtf(var + 1e-5f);
  const float vv[4] = {v.x, v.y, v.z, v.w};
  union { bf16 h[4]; uint2 u2; } p;
#pragma unroll
  for (int j = 0; j < 4; ++j) {
    const int c = tid * 4 + j;
    p.h[j] = (bf16)((vv[j] - mu) * rstd * g[c] + b[c]);
  }
  ((uint2*)(nrm + (size_t)row * DIM))[tid] = p.u2;
}

// ---------------------------------------------------------------------------
// Weight prep: W [Kd][Nd] f32 -> WT [Nd][Kd] bf16
// ---------------------------------------------------------------------------
__global__ void k_transpose_bf16(const float* __restrict__ W,
                                 bf16* __restrict__ WT, int Kd, int Nd)
{
  size_t idx = (size_t)blockIdx.x * 256 + threadIdx.x;
  if (idx >= (size_t)Kd * Nd) return;
  int n = (int)(idx / Kd);
  int k = (int)(idx % Kd);
  WT[idx] = (bf16)W[(size_t)k * Nd + n];
}

// ---------------------------------------------------------------------------
extern "C" void kernel_launch(void* const* d_in, const int* in_sizes, int n_in,
                              void* d_out, int out_size, void* d_ws, size_t ws_size,
                              hipStream_t stream)
{
  const float* x     = (const float*)d_in[0];
  const float* ln_g  = (const float*)d_in[1];
  const float* ln_b  = (const float*)d_in[2];
  const float* Wh    = (const float*)d_in[3];
  const float* bh    = (const float*)d_in[4];
  const float* Wqk   = (const float*)d_in[5];
  const float* bqk   = (const float*)d_in[6];
  const float* gamma = (const float*)d_in[7];
  const float* beta  = (const float*)d_in[8];
  const float* Wo    = (const float*)d_in[9];
  const float* bo    = (const float*)d_in[10];
  float* out = (float*)d_out;
  (void)in_sizes; (void)n_in; (void)out_size; (void)ws_size;

  char* ws = (char*)d_ws;
  size_t off = 0;
  auto take = [&](size_t bytes) {
    char* p = ws + off;
    off = (off + bytes + 255) & ~(size_t)255;
    return p;
  };
  bf16* nrm  = (bf16*)take((size_t)NTOK * DIM * 2);       // 16 MiB
  bf16* WhT  = (bf16*)take((size_t)2 * HID * DIM * 2);    //  8 MiB
  bf16* WqkT = (bf16*)take((size_t)QKD * DIM * 2);        // .25 MiB
  bf16* WoT  = (bf16*)take((size_t)DIM * HID * 2);        //  4 MiB
  bf16* vT   = (bf16*)take((size_t)HID * NTOK * 2);       // 32 MiB
  bf16* gate = (bf16*)take((size_t)NTOK * HID * 2);       // 32 MiB
  bf16* qb   = (bf16*)take((size_t)NTOK * QKD * 2);       //  2 MiB
  bf16* kb   = (bf16*)take((size_t)NTOK * QKD * 2);       //  2 MiB
  bf16* attn = (bf16*)take((size_t)2 * SEQ * SEQ * 2);    // 64 MiB
  bf16* u    = (bf16*)take((size_t)NTOK * HID * 2);       // 32 MiB

  // --- weight prep (bf16 transposed layouts) ---
  {
    size_t n1 = (size_t)DIM * 2 * HID;
    k_transpose_bf16<<<(unsigned)((n1 + 255) / 256), 256, 0, stream>>>(Wh, WhT, DIM, 2 * HID);
    size_t n2 = (size_t)DIM * QKD;
    k_transpose_bf16<<<(unsigned)((n2 + 255) / 256), 256, 0, stream>>>(Wqk, WqkT, DIM, QKD);
    size_t n3 = (size_t)HID * DIM;
    k_transpose_bf16<<<(unsigned)((n3 + 255) / 256), 256, 0, stream>>>(Wo, WoT, HID, DIM);
  }

  // --- LayerNorm ---
  k_layernorm<<<NTOK, 256, 0, stream>>>(x, ln_g, ln_b, nrm);

  // --- hv projection: SiLU -> vT / gate ---
  dim3 gHV(2 * HID / BN, NTOK / BM);           // (32, 64)
  k_gemm_hv<<<gHV, 128, 0, stream>>>(nrm, WhT, bh, vT, gate);

  // --- qk projection ---
  dim3 gQK(QKD / BN, NTOK / BM);               // (1, 64)
  k_gemm_qk<<<gQK, 128, 0, stream>>>(nrm, WqkT, bqk, gamma, beta, qb, kb);

  // --- attention per batch ---
  for (int b = 0; b < 2; ++b) {
    dim3 gS(SEQ / BN, SEQ / BM);               // (32, 32)
    k_gemm_sim<<<gS, 128, 0, stream>>>(qb + (size_t)b * SEQ * QKD,
                                       kb + (size_t)b * SEQ * QKD,
                                       attn + (size_t)b * SEQ * SEQ);
    dim3 gA(HID / BN, SEQ / BM);               // (16, 32)
    k_gemm_av<<<gA, 128, 0, stream>>>(attn + (size_t)b * SEQ * SEQ,
                                      vT + (size_t)b * SEQ,
                                      gate + (size_t)b * SEQ * HID,
                                      u + (size_t)b * SEQ * HID);
  }

  // --- output projection + residual ---
  dim3 gO(DIM / BN, NTOK / BM);                // (8, 64)
  k_gemm_out<<<gO, 128, 0, stream>>>(u, WoT, bo, x, out);
}
